// MultiHeadAttentionLayer_10651518894805
// MI455X (gfx1250) — matrix operand order repaired and checked by reading, not compiled, not run on hardware
//
#include <hip/hip_runtime.h>

#define B_   2
#define S_   2048
#define DIM_ 1024
#define H_   16
#define DH_  64

typedef __attribute__((ext_vector_type(16))) __bf16 v16bf;
typedef __attribute__((ext_vector_type(8)))  __bf16 v8bf;
typedef __attribute__((ext_vector_type(4)))  __bf16 v4bf;
typedef __attribute__((ext_vector_type(8)))  float  v8f;

// ---------------------------------------------------------------------------
// WMMA wrapper: D(16x16 f32) = A(16x32 bf16) * B(32x16 bf16) + C
// ---------------------------------------------------------------------------
__device__ __forceinline__ v8f wmma_bf16(v16bf a, v16bf b, v8f c) {
  return __builtin_amdgcn_wmma_f32_16x16x32_bf16(
      /*neg_a=*/false, a, /*neg_b=*/false, b,
      /*c_mod=*/(short)0, c, /*reuse_a=*/false, /*reuse_b=*/false);
}

// ---------------------------------------------------------------------------
// Fragment loaders (wave32; this code uses contiguous k order for A and B, not ISA table order):
//  A (16x32, 16-bit): lane holds row M=lane&15; elem i -> K = i + 16*hi
//    => 16 contiguous K values at col0+16*hi (matches B).
//  B (32x16, 16-bit): lane holds col N=lane&15; elem i -> K = i + 16*hi
//    => 16 contiguous K values when K-dim is contiguous in memory.
//  C (16x16, f32)   : elem r -> row = r + 8*hi, col = lane&15
// ---------------------------------------------------------------------------
__device__ __forceinline__ v16bf a_frag_bf16(const __bf16* p, int ld, int row,
                                             int col0, int hi) {
  const __bf16* r = p + (size_t)row * ld + col0 + 16 * hi;
  v8bf lo = *(const v8bf*)(r);        // 16B
  v8bf hv = *(const v8bf*)(r + 8);    // 16B
  return __builtin_shufflevector(lo, hv, 0, 1, 2, 3, 4, 5, 6, 7,
                                 8, 9, 10, 11, 12, 13, 14, 15);
}

// B fragment where K-dim is contiguous: element (k,n) at p[n*ldn + k]
__device__ __forceinline__ v16bf b_frag_bf16(const __bf16* p, int ldn, int n,
                                             int k0, int hi) {
  return *(const v16bf*)(p + (size_t)n * ldn + k0 + 16 * hi);  // 32B
}

// ---------------------------------------------------------------------------
// Prep kernel A: flat f32 -> bf16 convert (x and wo), 4 elems/thread
// ---------------------------------------------------------------------------
__global__ __launch_bounds__(256) void cvt_bf16_kernel(
    const float* __restrict__ in, __bf16* __restrict__ out, int n4) {
  int i = blockIdx.x * blockDim.x + threadIdx.x;
  if (i >= n4) return;
  float4 f = *(const float4*)(in + (size_t)i * 4);
  v4bf v;
  v[0] = (__bf16)f.x; v[1] = (__bf16)f.y; v[2] = (__bf16)f.z; v[3] = (__bf16)f.w;
  *(v4bf*)(out + (size_t)i * 4) = v;
}

// ---------------------------------------------------------------------------
// Prep kernel B: transpose-convert w[H][DIM][DH] f32 -> wT[H][DH][DIM] bf16.
//  blockIdx = (d, h, which); thread = f. Reads coalesced over f.
// ---------------------------------------------------------------------------
__global__ __launch_bounds__(64) void wtrans_kernel(
    const float* __restrict__ wq, const float* __restrict__ wk,
    const float* __restrict__ wv, __bf16* __restrict__ wqT,
    __bf16* __restrict__ wkT, __bf16* __restrict__ wvT) {
  const int f = threadIdx.x;      // 0..63
  const int d = blockIdx.x;       // 0..1023
  const int h = blockIdx.y;       // 0..15
  const float* w;
  __bf16* wT;
  if (blockIdx.z == 0)      { w = wq; wT = wqT; }
  else if (blockIdx.z == 1) { w = wk; wT = wkT; }
  else                      { w = wv; wT = wvT; }
  wT[((size_t)h * DH_ + f) * DIM_ + d] = (__bf16)w[((size_t)h * DIM_ + d) * DH_ + f];
}

// ---------------------------------------------------------------------------
// Kernel 1: fused QKV projection (all operands bf16, vector loads only).
//  Wave: 32 rows x 64 cols of q,k,v each (2 A-frags reuse every B-frag).
//  q/k stored row-major [B,H,S,DH]; v stored transposed [B,H,DH,S].
// ---------------------------------------------------------------------------
__global__ __launch_bounds__(128) void qkv_proj_kernel(
    const __bf16* __restrict__ xb,
    const __bf16* __restrict__ wqT, const __bf16* __restrict__ wkT,
    const __bf16* __restrict__ wvT,
    const float* __restrict__ bq, const float* __restrict__ bk,
    const float* __restrict__ bv,
    __bf16* __restrict__ qb, __bf16* __restrict__ kb, __bf16* __restrict__ vTb) {
  const int lane = threadIdx.x & 31;
  const int wave = threadIdx.x >> 5;
  const int l    = lane & 15;
  const int hi   = lane >> 4;

  const int rowtiles = S_ / 128;
  const int b    = blockIdx.x / rowtiles;
  const int rt   = blockIdx.x % rowtiles;
  const int h    = blockIdx.y;
  const int row0 = rt * 128 + wave * 32;

  const __bf16* X  = xb + (size_t)b * S_ * DIM_;
  const __bf16* Wq = wqT + (size_t)h * DH_ * DIM_;
  const __bf16* Wk = wkT + (size_t)h * DH_ * DIM_;
  const __bf16* Wv = wvT + (size_t)h * DH_ * DIM_;

  v8f cq[2][4], ck[2][4], cv[2][4];
#pragma unroll
  for (int u = 0; u < 2; u++)
#pragma unroll
    for (int t = 0; t < 4; t++) {
      cq[u][t] = (v8f)0.0f; ck[u][t] = (v8f)0.0f; cv[u][t] = (v8f)0.0f;
    }

  for (int k0 = 0; k0 < DIM_; k0 += 32) {
    v16bf a0 = a_frag_bf16(X, DIM_, row0 + l, k0, hi);
    v16bf a1 = a_frag_bf16(X, DIM_, row0 + 16 + l, k0, hi);
#pragma unroll
    for (int t = 0; t < 4; t++) {
      v16bf bfq = b_frag_bf16(Wq, DIM_, t * 16 + l, k0, hi);
      cq[0][t] = wmma_bf16(a0, bfq, cq[0][t]);
      cq[1][t] = wmma_bf16(a1, bfq, cq[1][t]);
      v16bf bfk = b_frag_bf16(Wk, DIM_, t * 16 + l, k0, hi);
      ck[0][t] = wmma_bf16(a0, bfk, ck[0][t]);
      ck[1][t] = wmma_bf16(a1, bfk, ck[1][t]);
      v16bf bfv = b_frag_bf16(Wv, DIM_, t * 16 + l, k0, hi);
      cv[0][t] = wmma_bf16(a0, bfv, cv[0][t]);
      cv[1][t] = wmma_bf16(a1, bfv, cv[1][t]);
    }
  }

  __bf16* qo = qb  + (((size_t)b * H_ + h) * S_ + row0) * DH_;
  __bf16* ko = kb  + (((size_t)b * H_ + h) * S_ + row0) * DH_;
  __bf16* vo = vTb + ((size_t)b * H_ + h) * DH_ * S_;
#pragma unroll
  for (int u = 0; u < 2; u++) {
#pragma unroll
    for (int t = 0; t < 4; t++) {
      int col = t * 16 + l;
      float biq = bq[h * DH_ + col];
      float bik = bk[h * DH_ + col];
      float biv = bv[h * DH_ + col];
#pragma unroll
      for (int r = 0; r < 8; r++) {
        int row = u * 16 + r + 8 * hi;
        qo[(size_t)row * DH_ + col] = (__bf16)((cq[u][t][r] + biq) * 0.125f);
        ko[(size_t)row * DH_ + col] = (__bf16)(ck[u][t][r] + bik);
      }
      // v transposed: fixed col, 8 consecutive rows -> one 16B store
      v8bf pk;
#pragma unroll
      for (int r = 0; r < 8; r++) pk[r] = (__bf16)(cv[u][t][r] + biv);
      *(v8bf*)(vo + (size_t)col * S_ + row0 + u * 16 + 8 * hi) = pk;
    }
  }
}

// ---------------------------------------------------------------------------
// Kernel 2: flash attention per (b, h, 64-row q tile); wave owns 16 q rows.
//  Online softmax over causal 32-wide key tiles; P converted C->A layout via
//  per-wave private LDS tile (same-wave dependency, s_wait_dscnt only).
// ---------------------------------------------------------------------------
__global__ __launch_bounds__(128) void flash_attn_kernel(
    const __bf16* __restrict__ qb, const __bf16* __restrict__ kb,
    const __bf16* __restrict__ vTb, const int* __restrict__ ignore,
    __bf16* __restrict__ ao) {
  __shared__ __bf16 pbuf[4][16][32];

  const int lane = threadIdx.x & 31;
  const int wave = threadIdx.x >> 5;
  const int l    = lane & 15;
  const int hi   = lane >> 4;

  const int rowtiles = S_ / 64;
  const int b    = blockIdx.x / rowtiles;
  const int rt   = blockIdx.x % rowtiles;
  const int h    = blockIdx.y;
  const int row0 = rt * 64 + wave * 16;

  const __bf16* Q  = qb  + ((size_t)b * H_ + h) * S_ * DH_;
  const __bf16* K  = kb  + ((size_t)b * H_ + h) * S_ * DH_;
  const __bf16* VT = vTb + ((size_t)b * H_ + h) * DH_ * S_;
  const int*    ig = ignore + (size_t)b * S_;

  // Q A-fragments for this wave's 16 rows (feature halves 0..31, 32..63)
  v16bf aq0 = a_frag_bf16(Q, DH_, row0 + l, 0, hi);
  v16bf aq1 = a_frag_bf16(Q, DH_, row0 + l, 32, hi);

  float m[8], ls[8];
  v8f   o[4];
#pragma unroll
  for (int r = 0; r < 8; r++) { m[r] = -3.0e38f; ls[r] = 0.0f; }
#pragma unroll
  for (int t = 0; t < 4; t++) o[t] = (v8f)0.0f;

  const int nkt = ((row0 + 15) >> 5) + 1;  // causal tile count
  for (int kt = 0; kt < nkt; kt++) {
    const int k0  = kt * 32;
    const int kp0 = k0 + l;
    const int kp1 = k0 + 16 + l;

    // S = Q K^T  (two 16x16 score tiles; K rows contiguous over features)
    v8f s0 = (v8f)0.0f, s1 = (v8f)0.0f;
    {
      v16bf b00 = b_frag_bf16(K, DH_, kp0, 0, hi);
      v16bf b01 = b_frag_bf16(K, DH_, kp0, 32, hi);
      s0 = wmma_bf16(aq0, b00, s0);
      s0 = wmma_bf16(aq1, b01, s0);
      v16bf b10 = b_frag_bf16(K, DH_, kp1, 0, hi);
      v16bf b11 = b_frag_bf16(K, DH_, kp1, 32, hi);
      s1 = wmma_bf16(aq0, b10, s1);
      s1 = wmma_bf16(aq1, b11, s1);
    }

    const int ig0 = ig[kp0];
    const int ig1 = ig[kp1];

    // mask + online softmax (row r of C frag = q row row0 + r + 8*hi)
#pragma unroll
    for (int r = 0; r < 8; r++) {
      const int qrow = row0 + r + 8 * hi;
      float v0 = s0[r], v1 = s1[r];
      bool ok0 = (kp0 <= qrow) && ((ig0 == 0) || (kp0 == qrow));
      bool ok1 = (kp1 <= qrow) && ((ig1 == 0) || (kp1 == qrow));
      v0 = ok0 ? v0 : -3.0e38f;
      v1 = ok1 ? v1 : -3.0e38f;
      s0[r] = v0;
      s1[r] = v1;

      float mx = fmaxf(v0, v1);
      mx = fmaxf(mx, __shfl_xor(mx, 1, 32));
      mx = fmaxf(mx, __shfl_xor(mx, 2, 32));
      mx = fmaxf(mx, __shfl_xor(mx, 4, 32));
      mx = fmaxf(mx, __shfl_xor(mx, 8, 32));

      float mn    = fmaxf(m[r], mx);
      float alpha = __expf(m[r] - mn);  // both -3e38 -> exp(0)=1, harmless
      float p0 = (s0[r] < -1.0e30f) ? 0.0f : __expf(s0[r] - mn);
      float p1 = (s1[r] < -1.0e30f) ? 0.0f : __expf(s1[r] - mn);
      s0[r] = p0;
      s1[r] = p1;

      float rs = p0 + p1;
      rs += __shfl_xor(rs, 1, 32);
      rs += __shfl_xor(rs, 2, 32);
      rs += __shfl_xor(rs, 4, 32);
      rs += __shfl_xor(rs, 8, 32);

      ls[r] = ls[r] * alpha + rs;
      m[r]  = mn;
#pragma unroll
      for (int t = 0; t < 4; t++) o[t][r] *= alpha;
    }

    // P: C layout -> LDS (row-major 16x32) -> A layout (2x ds_load_b128)
#pragma unroll
    for (int r = 0; r < 8; r++) {
      pbuf[wave][r + 8 * hi][l]      = (__bf16)s0[r];
      pbuf[wave][r + 8 * hi][16 + l] = (__bf16)s1[r];
    }
    asm volatile("s_wait_dscnt 0" ::: "memory");
    v16bf pa = a_frag_bf16(&pbuf[wave][0][0], 32, l, 0, hi);

    // O += P V  (V^T layout: K-dim = key position contiguous -> 32B loads)
#pragma unroll
    for (int t = 0; t < 4; t++) {
      v16bf bv = b_frag_bf16(VT, S_, t * 16 + l, k0, hi);
      o[t] = wmma_bf16(pa, bv, o[t]);
    }
    asm volatile("s_wait_dscnt 0" ::: "memory");  // pa consumed before next stores
  }

  // normalize and store head-concatenated attention output [B,S,DIM] bf16
  __bf16* aout = ao + ((size_t)b * S_ + row0) * DIM_ + h * DH_;
#pragma unroll
  for (int r = 0; r < 8; r++) {
    float inv = 1.0f / ls[r];
    int row = r + 8 * hi;
#pragma unroll
    for (int t = 0; t < 4; t++)
      aout[(size_t)row * DIM_ + t * 16 + l] = (__bf16)(o[t][r] * inv);
  }
}

// ---------------------------------------------------------------------------
// Kernel 3: output projection  out[m,d] = ao[m,:] @ wo[d,:] + bo[d]  (f32 out)
//  Wave: 32 rows x 64 cols (2 A-frags reuse every B-frag).
//  wo bf16 [d_out, d_in] row-major => B fragments are contiguous 32B loads.
// ---------------------------------------------------------------------------
__global__ __launch_bounds__(128) void out_proj_kernel(
    const __bf16* __restrict__ ao, const __bf16* __restrict__ wob,
    const float* __restrict__ bo, float* __restrict__ out) {
  const int lane = threadIdx.x & 31;
  const int wave = threadIdx.x >> 5;
  const int l    = lane & 15;
  const int hi   = lane >> 4;

  const int row0 = blockIdx.x * 128 + wave * 32;  // over B*S rows
  const int col0 = blockIdx.y * 64;               // over DIM cols

  v8f c[2][4];
#pragma unroll
  for (int u = 0; u < 2; u++)
#pragma unroll
    for (int t = 0; t < 4; t++) c[u][t] = (v8f)0.0f;

  for (int k0 = 0; k0 < DIM_; k0 += 32) {
    v16bf a0 = a_frag_bf16(ao, DIM_, row0 + l, k0, hi);
    v16bf a1 = a_frag_bf16(ao, DIM_, row0 + 16 + l, k0, hi);
#pragma unroll
    for (int t = 0; t < 4; t++) {
      v16bf bf = b_frag_bf16(wob, DIM_, col0 + t * 16 + l, k0, hi);
      c[0][t] = wmma_bf16(a0, bf, c[0][t]);
      c[1][t] = wmma_bf16(a1, bf, c[1][t]);
    }
  }

#pragma unroll
  for (int u = 0; u < 2; u++)
#pragma unroll
    for (int t = 0; t < 4; t++) {
      int col = col0 + t * 16 + l;
      float bias = bo[col];
#pragma unroll
      for (int r = 0; r < 8; r++) {
        int row = row0 + u * 16 + r + 8 * hi;
        out[(size_t)row * DIM_ + col] = c[u][t][r] + bias;
      }
    }
}

// ---------------------------------------------------------------------------
// Launcher
// ---------------------------------------------------------------------------
extern "C" void kernel_launch(void* const* d_in, const int* in_sizes, int n_in,
                              void* d_out, int out_size, void* d_ws,
                              size_t ws_size, hipStream_t stream) {
  const float* x      = (const float*)d_in[0];
  const int*   ignore = (const int*)d_in[1];
  const float* wq     = (const float*)d_in[2];
  const float* bq     = (const float*)d_in[3];
  const float* wk     = (const float*)d_in[4];
  const float* bk     = (const float*)d_in[5];
  const float* wv     = (const float*)d_in[6];
  const float* bv     = (const float*)d_in[7];
  const float* wo     = (const float*)d_in[8];
  const float* bo     = (const float*)d_in[9];

  const size_t nx   = (size_t)B_ * S_ * DIM_;      // 4M
  const size_t nw   = (size_t)H_ * DH_ * DIM_;     // 1M per proj weight
  const size_t nqkv = (size_t)B_ * H_ * S_ * DH_;  // 4M per q/k/v

  __bf16* xb  = (__bf16*)d_ws;
  __bf16* wqT = xb  + nx;
  __bf16* wkT = wqT + nw;
  __bf16* wvT = wkT + nw;
  __bf16* wob = wvT + nw;
  __bf16* qb  = wob + nw;
  __bf16* kb  = qb  + nqkv;
  __bf16* vTb = kb  + nqkv;       // [B,H,DH,S]
  __bf16* aoB = vTb + nqkv;       // [B,S,DIM]

  dim3 blk(128);

  // prep: convert x and wo; transpose-convert wq/wk/wv
  cvt_bf16_kernel<<<dim3((unsigned)(nx / 4 / 256)), dim3(256), 0, stream>>>(
      x, xb, (int)(nx / 4));
  cvt_bf16_kernel<<<dim3((unsigned)(nw / 4 / 256)), dim3(256), 0, stream>>>(
      wo, wob, (int)(nw / 4));
  wtrans_kernel<<<dim3(DIM_, H_, 3), dim3(64), 0, stream>>>(wq, wk, wv, wqT,
                                                            wkT, wvT);

  dim3 g1(B_ * (S_ / 128), H_);
  qkv_proj_kernel<<<g1, blk, 0, stream>>>(xb, wqT, wkT, wvT, bq, bk, bv, qb,
                                          kb, vTb);
  dim3 g2(B_ * (S_ / 64), H_);
  flash_attn_kernel<<<g2, blk, 0, stream>>>(qb, kb, vTb, ignore, aoB);
  dim3 g3((B_ * S_) / 128, DIM_ / 64);
  out_proj_kernel<<<g3, blk, 0, stream>>>(aoB, wob, bo, (float*)d_out);
}
